// GCN_LSTM_44306882626193
// MI455X (gfx1250) — compile-verified, matrix operand order checked by reading, not run on hardware
//
#include <hip/hip_runtime.h>
#include <hip/hip_bf16.h>

// ---------------- problem constants ----------------
#define BB 4096   // batch
#define TT 12     // time steps
#define NN 132    // nodes
#define FF 19     // features
#define SS 4      // static features
#define DD 23     // GRU input size (F+S), padded to 32
#define GU 50     // GRU units, padded to 64
#define G3 150    // 3*GRU units, padded to 160 (10 chunks of 16)

// packed sizes (bf16 elements)
#define N_ADJ (144 * 160)          // adj A-matrix, row-major padded
#define N_KPK (2 * 512)            // gcn_kernel B-fragments: 2 col-chunks
#define N_WPK (NN * 10 * 512)      // W B-fragments: per node, 10 col-chunks, 1 k-step
#define N_UPK (NN * 2 * 10 * 512)  // U B-fragments: per node, 2 k-steps, 10 col-chunks
#define N_AL  ((size_t)NN * BB * TT * 32)  // al[n][b][t][32] bf16

typedef __attribute__((ext_vector_type(16))) __bf16 v16bf;
typedef __attribute__((ext_vector_type(8)))  float  v8f;

union FragU { uint4 q[2]; v16bf v; };

// K index inside a 32-wide k-step for fragment element i, lane-half hi
__device__ __forceinline__ int kof_i(int i, int hi) {
  return (i < 8) ? (hi * 8 + i) : (16 + hi * 8 + (i - 8));
}

// A-matrix 16x32 bf16 fragment from row-major storage (rowstride in halfs).
// Lane l: m = l&15, hi = l>>4; needs K runs [hi*8, hi*8+7] and [16+hi*8, ...].
// Two 16-byte contiguous loads per lane.
__device__ __forceinline__ v16bf load_a_frag(const __bf16* base, int rowstride) {
  int lane = threadIdx.x & 31;
  int m = lane & 15, hi = lane >> 4;
  const __bf16* p = base + m * rowstride + hi * 8;
  FragU u;
  u.q[0] = *(const uint4*)(p);
  u.q[1] = *(const uint4*)(p + 16);
  return u.v;
}

// B-matrix 32x16 bf16 fragment, pre-packed lane-major (32 lanes x 16 halfs).
__device__ __forceinline__ v16bf load_b_frag(const __bf16* blk) {
  const __bf16* p = blk + (threadIdx.x & 31) * 16;
  FragU u;
  u.q[0] = *(const uint4*)(p);
  u.q[1] = *(const uint4*)(p + 8);
  return u.v;
}

__device__ __forceinline__ v8f wmma_bf16(v16bf a, v16bf b, v8f c) {
  return __builtin_amdgcn_wmma_f32_16x16x32_bf16(
      /*neg_a=*/false, a, /*neg_b=*/false, b, /*c_mod=*/(short)0, c,
      /*reuse_a=*/false, /*reuse_b=*/false);
}

// ---------------------------------------------------------------------------
// Kernel 1: convert/pack weights to bf16 fragment layouts.
// ---------------------------------------------------------------------------
__global__ void __launch_bounds__(256) prep_kernel(
    const float* __restrict__ adj, const float* __restrict__ gk,
    const float* __restrict__ W, const float* __restrict__ Uw,
    __bf16* __restrict__ adjb, __bf16* __restrict__ kpk,
    __bf16* __restrict__ wpk, __bf16* __restrict__ upk) {
  int idx = blockIdx.x * 256 + threadIdx.x;
  if (idx < N_ADJ) {  // adj: row-major 144x160, zero padded
    int m = idx / 160, k = idx % 160;
    float v = (m < NN && k < NN) ? adj[m * NN + k] : 0.f;
    adjb[idx] = (__bf16)v;
    return;
  }
  idx -= N_ADJ;
  if (idx < N_KPK) {  // gcn_kernel: B-fragment pack (K=19->32, cols 19->32)
    int c = idx >> 9, r = idx & 511;
    int lane = r >> 4, i = r & 15, hi = lane >> 4;
    int K = kof_i(i, hi);
    int col = c * 16 + (lane & 15);
    float v = (K < FF && col < FF) ? gk[K * FF + col] : 0.f;
    kpk[idx] = (__bf16)v;
    return;
  }
  idx -= N_KPK;
  if (idx < N_WPK) {  // W[n][23][150] -> per node 10 chunk blocks (K=32)
    int n = idx / 5120, rem = idx % 5120;
    int c = rem >> 9, r = rem & 511;
    int lane = r >> 4, i = r & 15, hi = lane >> 4;
    int K = kof_i(i, hi);
    int col = c * 16 + (lane & 15);
    float v = (K < DD && col < G3) ? W[((size_t)n * DD + K) * G3 + col] : 0.f;
    wpk[idx] = (__bf16)v;
    return;
  }
  idx -= N_WPK;
  if (idx < N_UPK) {  // U[n][50][150] -> per node 2 k-steps x 10 chunks
    int n = idx / 10240, rem = idx % 10240;
    int s = rem / 5120, rem2 = rem % 5120;
    int c = rem2 >> 9, r = rem2 & 511;
    int lane = r >> 4, i = r & 15, hi = lane >> 4;
    int K = s * 32 + kof_i(i, hi);
    int col = c * 16 + (lane & 15);
    float v = (K < GU && col < G3) ? Uw[((size_t)n * GU + K) * G3 + col] : 0.f;
    upk[idx] = (__bf16)v;
  }
}

// ---------------------------------------------------------------------------
// Kernel 2: GCN. One WG per (b,t): al[:,b,t,:] = concat(relu(adj@temp@K + b), stat)
// GEMM1: A=adj(144x160 bf16, global), B=temp_bt packed in LDS -> g (LDS bf16)
// GEMM2: A=g, B=gcn_kernel packed (global) -> relu -> al (bf16 global)
// ---------------------------------------------------------------------------
__global__ void __launch_bounds__(256) gcn_kernel(
    const float* __restrict__ temp, const float* __restrict__ stat,
    const float* __restrict__ gbias, const __bf16* __restrict__ adjb,
    const __bf16* __restrict__ kpk, __bf16* __restrict__ alb) {
  __shared__ __align__(16) __bf16 sB[5 * 2 * 512];  // temp_bt B-fragments
  __shared__ __align__(16) __bf16 sG[144 * 32];     // g, row-major A layout

  int b = blockIdx.x / TT, t = blockIdx.x % TT;
  int tid = threadIdx.x;
  int wave = tid >> 5;
  const float* tbt = temp + ((size_t)b * TT + t) * NN * FF;

  // pack temp_bt (K=node 132->160, col=f 19->32) into B-fragment order
  for (int idx = tid; idx < 5 * 2 * 512; idx += 256) {
    int s = idx / 1024;
    int c = (idx >> 9) & 1;
    int r = idx & 511;
    int lane = r >> 4, i = r & 15, hi = lane >> 4;
    int K = s * 32 + kof_i(i, hi);
    int col = c * 16 + (lane & 15);
    float v = (K < NN && col < FF) ? tbt[K * FF + col] : 0.f;
    sB[idx] = (__bf16)v;
  }
  __syncthreads();

  // GEMM1: g = adj @ temp_bt   (9 m-tiles x 2 n-chunks, 5 k-steps)
  for (int tile = wave; tile < 18; tile += 8) {
    int mt = tile >> 1, c = tile & 1;
    v8f acc = {};
    for (int s = 0; s < 5; ++s) {
      v16bf a = load_a_frag(adjb + mt * 16 * 160 + s * 32, 160);
      v16bf bfr = load_b_frag(sB + (s * 2 + c) * 512);
      acc = wmma_bf16(a, bfr, acc);
    }
    int lane = tid & 31, hi = lane >> 4, nlo = lane & 15;
    for (int r = 0; r < 8; ++r)
      sG[(mt * 16 + hi * 8 + r) * 32 + c * 16 + nlo] = (__bf16)acc[r];
  }
  __syncthreads();

  // GEMM2: X = relu(g @ K19 + bias); store valid region of al
  for (int tile = wave; tile < 18; tile += 8) {
    int mt = tile >> 1, c = tile & 1;
    v16bf a = load_a_frag(sG + mt * 16 * 32, 32);
    v16bf bfr = load_b_frag(kpk + c * 512);
    v8f acc = {};
    acc = wmma_bf16(a, bfr, acc);
    int lane = tid & 31, hi = lane >> 4, nlo = lane & 15;
    int col = c * 16 + nlo;
    float bias = (col < FF) ? gbias[col] : 0.f;
    for (int r = 0; r < 8; ++r) {
      int m = mt * 16 + hi * 8 + r;
      if (m < NN && col < FF) {
        float v = acc[r] + bias;
        v = v > 0.f ? v : 0.f;
        alb[(((size_t)m * BB + b) * TT + t) * 32 + col] = (__bf16)v;
      }
    }
  }

  // static feature concat (cols 19..22) + zero pad (cols 23..31)
  for (int idx = tid; idx < NN * 13; idx += 256) {
    int node = idx / 13, col = 19 + idx % 13;
    float v = (col < DD) ? stat[((size_t)b * NN + node) * SS + (col - 19)] : 0.f;
    alb[(((size_t)node * BB + b) * TT + t) * 32 + col] = (__bf16)v;
  }
}

// ---------------------------------------------------------------------------
// Kernel 3: grouped GRU + dense + ELU + transpose.
// One WG per (node, 16-row batch tile). 10 waves = 10 column chunks of 160.
// Weight B-fragments are t-invariant -> hoisted into VGPRs before the scan.
// ---------------------------------------------------------------------------
__global__ void __launch_bounds__(320) gru_kernel(
    const float* __restrict__ b_in, const float* __restrict__ b_rec,
    const float* __restrict__ Wd, const float* __restrict__ bd,
    const __bf16* __restrict__ wpk, const __bf16* __restrict__ upk,
    const __bf16* __restrict__ alb, float* __restrict__ out) {
  __shared__ __align__(16) __bf16 sX[16 * 32];   // x_t, A layout (K=32)
  __shared__ __align__(16) __bf16 sHb[16 * 64];  // h bf16, A layout (K=64 padded)
  __shared__ float sHf[16 * 50];                 // h fp32
  __shared__ float sAx[16 * 160];                // x@W accumulators
  __shared__ float sAu[16 * 160];                // h@U accumulators

  int n = blockIdx.x % NN;
  int b0 = (blockIdx.x / NN) * 16;
  int tid = threadIdx.x;
  int wave = tid >> 5;
  int lane = tid & 31;

  // hoist per-node weight fragments (reused all 12 steps, L2-hot)
  v16bf wf  = load_b_frag(wpk + (size_t)n * 5120 + wave * 512);
  v16bf uf0 = load_b_frag(upk + (size_t)n * 10240 + wave * 512);
  v16bf uf1 = load_b_frag(upk + (size_t)n * 10240 + 5120 + wave * 512);

  for (int i = tid; i < 16 * 50; i += 320) sHf[i] = 0.f;
  for (int i = tid; i < 16 * 64; i += 320) sHb[i] = (__bf16)0.f;
  __syncthreads();

  const unsigned int* alu = (const unsigned int*)alb;
  for (int t = 0; t < TT; ++t) {
    // stage x_t tile: 16 rows x 32 halfs (as 256 dwords)
    if (tid < 256) {
      int row = tid >> 4, k2 = tid & 15;
      ((unsigned int*)sX)[tid] =
          alu[(((size_t)n * BB + b0 + row) * TT + t) * 16 + k2];
    }
    __syncthreads();
    {
      v16bf ax = load_a_frag(sX, 32);
      v8f accx = {};
      accx = wmma_bf16(ax, wf, accx);            // x_t @ W  (K=32)
      v16bf ah0 = load_a_frag(sHb, 64);
      v16bf ah1 = load_a_frag(sHb + 32, 64);
      v8f accu = {};
      accu = wmma_bf16(ah0, uf0, accu);          // h @ U, K=0..31
      accu = wmma_bf16(ah1, uf1, accu);          // h @ U, K=32..63
      int hi = lane >> 4, nlo = lane & 15;
      int col = wave * 16 + nlo;
      for (int r = 0; r < 8; ++r) {
        sAx[(hi * 8 + r) * 160 + col] = accx[r];
        sAu[(hi * 8 + r) * 160 + col] = accu[r];
      }
    }
    __syncthreads();
    // gates (Keras reset_after, order z,r,h)
    for (int idx = tid; idx < 16 * 50; idx += 320) {
      int row = idx / 50, u = idx % 50;
      float xz = sAx[row * 160 + u]       + b_in[n * G3 + u];
      float xr = sAx[row * 160 + u + 50]  + b_in[n * G3 + u + 50];
      float xh = sAx[row * 160 + u + 100] + b_in[n * G3 + u + 100];
      float rz = sAu[row * 160 + u]       + b_rec[n * G3 + u];
      float rr = sAu[row * 160 + u + 50]  + b_rec[n * G3 + u + 50];
      float rh = sAu[row * 160 + u + 100] + b_rec[n * G3 + u + 100];
      float z = 1.f / (1.f + __expf(-(xz + rz)));
      float r = 1.f / (1.f + __expf(-(xr + rr)));
      float hh = tanhf(xh + r * rh);
      float hn = z * sHf[idx] + (1.f - z) * hh;
      sHf[idx] = hn;
      sHb[row * 64 + u] = (__bf16)hn;
    }
    __syncthreads();
  }

  // dense + ELU, store transposed [B,H,N]
  if (tid < 48) {
    int row = tid / 3, h = tid % 3;
    float acc = bd[n * 3 + h];
    for (int u = 0; u < GU; ++u)
      acc += sHf[row * 50 + u] * Wd[((size_t)n * GU + u) * 3 + h];
    float e = acc > 0.f ? acc : (__expf(acc) - 1.f);
    out[((size_t)(b0 + row) * 3 + h) * NN + n] = e;
  }
}

// ---------------------------------------------------------------------------
extern "C" void kernel_launch(void* const* d_in, const int* in_sizes, int n_in,
                              void* d_out, int out_size, void* d_ws, size_t ws_size,
                              hipStream_t stream) {
  const float* temp  = (const float*)d_in[0];
  const float* stat  = (const float*)d_in[1];
  const float* adj   = (const float*)d_in[2];
  const float* gk    = (const float*)d_in[3];
  const float* gbias = (const float*)d_in[4];
  const float* W     = (const float*)d_in[5];
  const float* Uw    = (const float*)d_in[6];
  const float* b_in  = (const float*)d_in[7];
  const float* b_rec = (const float*)d_in[8];
  const float* Wd    = (const float*)d_in[9];
  const float* bd    = (const float*)d_in[10];
  float* out = (float*)d_out;

  __bf16* adjb = (__bf16*)d_ws;            // 23040 halfs
  __bf16* kpk  = adjb + N_ADJ;             // 1024
  __bf16* wpk  = kpk + N_KPK;              // 675840
  __bf16* upk  = wpk + N_WPK;              // 1351680
  __bf16* alb  = upk + N_UPK;              // ~208M halfs (~396 MB)

  int prep_total = N_ADJ + N_KPK + N_WPK + N_UPK;
  prep_kernel<<<(prep_total + 255) / 256, 256, 0, stream>>>(
      adj, gk, W, Uw, adjb, kpk, wpk, upk);

  gcn_kernel<<<BB * TT, 256, 0, stream>>>(temp, stat, gbias, adjb, kpk, alb);

  gru_kernel<<<NN * (BB / 16), 320, 0, stream>>>(
      b_in, b_rec, Wd, bd, wpk, upk, alb, out);
}
